// GCNN_85409719648958
// MI455X (gfx1250) — matrix operand order repaired and checked by reading, not compiled, not run on hardware
//
#include <hip/hip_runtime.h>
#include <hip/hip_bf16.h>
#include <math.h>

// ---------- problem constants ----------
#define BB 8
#define NN 2000
#define EE 64000
#define FF 1024
#define DD 128
#define BN (BB * NN)          // 16000 rows
#define NEG_SLOPE 0.01f

// ---------- CDNA5 vector types (POD, safe in unions) ----------
typedef __bf16 bf16_t;
typedef bf16_t v16bf __attribute__((ext_vector_type(16)));
typedef float  v8f   __attribute__((ext_vector_type(8)));
typedef unsigned int u32x4 __attribute__((ext_vector_type(4)));

union Frag { u32x4 u[2]; v16bf v; };

// ---------- helpers ----------
__device__ __forceinline__ unsigned int pack_bf16_pair(float a, float b) {
    unsigned int ua = __float_as_uint(a);
    unsigned int ub = __float_as_uint(b);
    ua += 0x7FFFu + ((ua >> 16) & 1u);   // RNE
    ub += 0x7FFFu + ((ub >> 16) & 1u);
    return (ua >> 16) | (ub & 0xFFFF0000u);
}

__device__ __forceinline__ float leaky(float v) {
    return v >= 0.0f ? v : NEG_SLOPE * v;
}

// ---------- 1) fp32 -> packed bf16 pairs (row-major preserved) ----------
__global__ void convert_x_kernel(const float* __restrict__ x,
                                 unsigned int* __restrict__ xb, long long n2) {
    long long i = (long long)blockIdx.x * blockDim.x + threadIdx.x;
    if (i >= n2) return;
    xb[i] = pack_bf16_pair(x[2 * i], x[2 * i + 1]);
}

// ---------- 2) pack W [F,F] fp32 -> WMMA B-fragment layout ----------
// Wpack index t = ((kt*F + c)*2 + half)*8 + j ; stores (W[k0,c], W[k0+1,c]),
// k0 = kt*32 + (j>>2)*16 + half*8 + 2*(j&3).  Lane l (col c=l&15, half=l>>4) then reads
// 8 consecutive dwords = two b128 loads.
__global__ void pack_w_kernel(const float* __restrict__ W,
                              unsigned int* __restrict__ Wp) {
    int t = blockIdx.x * blockDim.x + threadIdx.x;           // < 32*1024*2*8
    if (t >= 32 * FF * 2 * 8) return;
    int j  = t & 7;
    int h  = (t >> 3) & 1;
    int c  = (t >> 4) & (FF - 1);
    int kt = t >> 14;
    int k0 = kt * 32 + (j >> 2) * 16 + h * 8 + 2 * (j & 3);
    Wp[t] = pack_bf16_pair(W[(size_t)k0 * FF + c], W[(size_t)(k0 + 1) * FF + c]);
}

// ---------- 3) bf16 WMMA GEMM: H[BN,F] = Xbf[BN,F] * W[F,F] ----------
// block = 128 threads (4 waves); each wave -> 16x64 tile; grid (F/64, BN/64)
__global__ __launch_bounds__(128) void gemm_bf16_kernel(
        const unsigned short* __restrict__ A,   // bf16 row-major [BN,F]
        const unsigned int*  __restrict__ Bp,   // packed W fragments
        float* __restrict__ C) {
    const int lane = threadIdx.x & 31;
    const int wave = threadIdx.x >> 5;
    const int half = lane >> 4;
    const int l15  = lane & 15;

    const int m0 = blockIdx.y * 64 + wave * 16;
    const int n0 = blockIdx.x * 64;
    const int r  = m0 + l15;

    const u32x4* Arow  = (const u32x4*)(A + (size_t)r * FF);
    const u32x4* Bbase = (const u32x4*)Bp;

    v8f acc0 = {0,0,0,0,0,0,0,0};
    v8f acc1 = {0,0,0,0,0,0,0,0};
    v8f acc2 = {0,0,0,0,0,0,0,0};
    v8f acc3 = {0,0,0,0,0,0,0,0};

    for (int kk = 0; kk < FF / 32; ++kk) {
        __builtin_prefetch((const char*)(Arow + (kk + 8) * 4), 0, 3);
        Frag a;
        a.u[0] = Arow[kk * 4 + half];        // K = half*8 .. +7
        a.u[1] = Arow[kk * 4 + half + 2];    // K = 16+half*8 .. +7

        const int c0 = n0 + l15;
#pragma unroll
        for (int nf = 0; nf < 4; ++nf) {
            const int c = c0 + nf * 16;
            Frag bfr;
            size_t bi = ((size_t)kk * FF + c) * 4 + half * 2;  // uint4 index
            bfr.u[0] = Bbase[bi];
            bfr.u[1] = Bbase[bi + 1];
            v8f* accp = (nf == 0) ? &acc0 : (nf == 1) ? &acc1
                       : (nf == 2) ? &acc2 : &acc3;
            *accp = __builtin_amdgcn_wmma_f32_16x16x32_bf16(
                        false, a.v, false, bfr.v, (short)0, *accp, false, false);
        }
    }

    // C/D layout: col = l15, row = m0 + half*8 + i
#pragma unroll
    for (int nf = 0; nf < 4; ++nf) {
        const v8f accv = (nf == 0) ? acc0 : (nf == 1) ? acc1
                        : (nf == 2) ? acc2 : acc3;
        const int c = n0 + nf * 16 + l15;
#pragma unroll
        for (int i = 0; i < 8; ++i) {
            int row = m0 + half * 8 + i;
            C[(size_t)row * FF + c] = accv[i];
        }
    }
}

// ---------- 4) CSR build: count, scan, fill ----------
__global__ void zero_int_kernel(int* __restrict__ a, int n) {
    int i = blockIdx.x * blockDim.x + threadIdx.x;
    if (i < n) a[i] = 0;
}

__global__ void zero_float_kernel(float* __restrict__ a, int n) {
    int i = blockIdx.x * blockDim.x + threadIdx.x;
    if (i < n) a[i] = 0.0f;
}

__global__ void count_kernel(const int* __restrict__ ei, int* __restrict__ cnt) {
    int t = blockIdx.x * blockDim.x + threadIdx.x;   // < BB*EE
    if (t >= BB * EE) return;
    int b = t / EE, e = t % EE;
    int tgt = ei[(size_t)b * 2 * EE + EE + e];
    atomicAdd(&cnt[b * NN + tgt], 1);
}

__global__ void dinv_kernel(const int* __restrict__ cnt, float* __restrict__ dinv, int n) {
    int i = blockIdx.x * blockDim.x + threadIdx.x;
    if (i < n) dinv[i] = rsqrtf(1.0f + (float)cnt[i]);   // +1 self loop
}

// one thread per graph: exclusive scan of 2000 counts (trivial work)
__global__ void scan_kernel(const int* __restrict__ cnt, int* __restrict__ offs) {
    int b = threadIdx.x;
    if (b >= BB) return;
    int run = b * EE;
    for (int i = 0; i < NN; ++i) {
        offs[b * NN + i] = run;
        run += cnt[b * NN + i];
    }
}

__global__ void fill_kernel(const int* __restrict__ ei,
                            const int* __restrict__ offs,
                            int* __restrict__ cursor,
                            int* __restrict__ csr_src) {
    int t = blockIdx.x * blockDim.x + threadIdx.x;   // < BB*EE
    if (t >= BB * EE) return;
    int b = t / EE, e = t % EE;
    const int* eb = ei + (size_t)b * 2 * EE;
    int s = eb[e];
    int tg = eb[EE + e];
    int row = b * NN + tg;
    int pos = offs[row] + atomicAdd(&cursor[row], 1);
    csr_src[pos] = s;
}

// ---------- 5) gather SpMM fused with bias + leaky + mean ----------
// block per target row; 256 threads x float4 covers F=1024.  No atomics on
// the bulk gather path; mean accumulated with one atomicAdd per element.
__global__ __launch_bounds__(256) void spmm_mean_kernel(
        const float* __restrict__ H, const int* __restrict__ csr_src,
        const int* __restrict__ offs, const int* __restrict__ cnt,
        const float* __restrict__ dinv, const float* __restrict__ bias,
        float* __restrict__ gmean) {
    const int row = blockIdx.x;              // < BN
    const int b   = row / NN;
    const int i   = threadIdx.x;             // float4 index, FF/4 == 256

    const float dt = dinv[row];
    const float4* H4 = (const float4*)H;

    // self loop: dinv[t]*dinv[t] * H[row]
    float4 v = H4[(size_t)row * (FF / 4) + i];
    float nrm = dt * dt;
    float4 acc;
    acc.x = nrm * v.x; acc.y = nrm * v.y; acc.z = nrm * v.z; acc.w = nrm * v.w;

    const int start = offs[row];
    const int m = cnt[row];
    for (int k = 0; k < m; ++k) {
        int s = csr_src[start + k];
        float nr = dt * dinv[b * NN + s];
        float4 h = H4[((size_t)b * NN + s) * (FF / 4) + i];
        acc.x += nr * h.x; acc.y += nr * h.y;
        acc.z += nr * h.z; acc.w += nr * h.w;
    }

    const int f = i * 4;
    const float inv_n = 1.0f / (float)NN;
    float r0 = leaky(acc.x + bias[f + 0]) * inv_n;
    float r1 = leaky(acc.y + bias[f + 1]) * inv_n;
    float r2 = leaky(acc.z + bias[f + 2]) * inv_n;
    float r3 = leaky(acc.w + bias[f + 3]) * inv_n;
    atomicAdd(&gmean[b * FF + f + 0], r0);
    atomicAdd(&gmean[b * FF + f + 1], r1);
    atomicAdd(&gmean[b * FF + f + 2], r2);
    atomicAdd(&gmean[b * FF + f + 3], r3);
}

// ---------- 6) small dense FC: out[b, n(+off)] = act(in[b,:] @ W + bias) ----------
__global__ void fc_kernel(const float* __restrict__ in,
                          const float* __restrict__ W,
                          const float* __restrict__ bias,
                          float* __restrict__ out,
                          int Bn, int K, int Nout, int ldout, int colOff,
                          int do_leaky) {
    int t = blockIdx.x * blockDim.x + threadIdx.x;
    if (t >= Bn * Nout) return;
    int b = t / Nout, n = t % Nout;
    float acc = bias[n];
    const float* ip = in + (size_t)b * K;
    for (int k = 0; k < K; ++k)
        acc += ip[k] * W[(size_t)k * Nout + n];
    if (do_leaky) acc = leaky(acc);
    out[(size_t)b * ldout + colOff + n] = acc;
}

// ---------- 7) final: sigmoid(t2 @ Wo + bo) ----------
__global__ void head_out_kernel(const float* __restrict__ t2,
                                const float* __restrict__ Wo,
                                const float* __restrict__ bo,
                                float* __restrict__ out) {
    int b = threadIdx.x;
    if (b >= BB) return;
    float acc = bo[0];
    for (int k = 0; k < 64; ++k) acc += t2[b * 64 + k] * Wo[k];
    out[b] = 1.0f / (1.0f + __expf(-acc));
}

// ================= host launcher =================
extern "C" void kernel_launch(void* const* d_in, const int* in_sizes, int n_in,
                              void* d_out, int out_size, void* d_ws, size_t ws_size,
                              hipStream_t stream) {
    // inputs (setup_inputs order)
    const float* x[2]  = { (const float*)d_in[0], (const float*)d_in[2] };
    const int*   ei[2] = { (const int*)d_in[1],   (const int*)d_in[3]   };
    const float* Wg[2] = { (const float*)d_in[4], (const float*)d_in[8] };
    const float* bg[2] = { (const float*)d_in[5], (const float*)d_in[9] };
    const float* Wf[2] = { (const float*)d_in[6], (const float*)d_in[10]};
    const float* bf_[2]= { (const float*)d_in[7], (const float*)d_in[11]};
    const float* W1 = (const float*)d_in[12];
    const float* b1 = (const float*)d_in[13];
    const float* W2 = (const float*)d_in[14];
    const float* b2 = (const float*)d_in[15];
    const float* Wo = (const float*)d_in[16];
    const float* bo = (const float*)d_in[17];
    float* out = (float*)d_out;

    // workspace carve-up (256B aligned)
    char* p = (char*)d_ws;
    auto carve = [&](size_t bytes) -> void* {
        void* r = (void*)p;
        p += (bytes + 255) & ~(size_t)255;
        return r;
    };
    unsigned int* xb   = (unsigned int*)carve((size_t)BN * FF * 2);      // bf16 X
    unsigned int* Wp   = (unsigned int*)carve((size_t)32 * FF * 16 * 4); // packed W
    float* H     = (float*)carve((size_t)BN * FF * 4);
    float* dinv  = (float*)carve((size_t)BN * 4);
    int*   cnt   = (int*)carve((size_t)BN * 4);
    int*   offs  = (int*)carve((size_t)BN * 4);
    int*   curs  = (int*)carve((size_t)BN * 4);
    int*   csr   = (int*)carve((size_t)BB * EE * 4);
    float* gmean = (float*)carve((size_t)BB * FF * 4);
    float* xc    = (float*)carve((size_t)BB * 256 * 4);
    float* t1    = (float*)carve((size_t)BB * 256 * 4);
    float* t2    = (float*)carve((size_t)BB * 64 * 4);
    (void)ws_size; (void)in_sizes; (void)n_in; (void)out_size;

    for (int br = 0; br < 2; ++br) {
        // X -> bf16 ; W -> fragment-packed bf16
        long long n2 = (long long)BN * FF / 2;
        convert_x_kernel<<<(unsigned)((n2 + 255) / 256), 256, 0, stream>>>(x[br], xb, n2);
        pack_w_kernel<<<(32 * FF * 2 * 8) / 256, 256, 0, stream>>>(Wg[br], Wp);

        // H = X @ W  (WMMA bf16)
        gemm_bf16_kernel<<<dim3(FF / 64, BN / 64), 128, 0, stream>>>(
            (const unsigned short*)xb, Wp, H);

        // CSR of incoming edges + dinv
        zero_int_kernel<<<(BN + 255) / 256, 256, 0, stream>>>(cnt, BN);
        count_kernel<<<(BB * EE + 255) / 256, 256, 0, stream>>>(ei[br], cnt);
        dinv_kernel<<<(BN + 255) / 256, 256, 0, stream>>>(cnt, dinv, BN);
        scan_kernel<<<1, 32, 0, stream>>>(cnt, offs);
        zero_int_kernel<<<(BN + 255) / 256, 256, 0, stream>>>(curs, BN);
        fill_kernel<<<(BB * EE + 255) / 256, 256, 0, stream>>>(ei[br], offs, curs, csr);

        // gather SpMM fused with self-loop, bias, leaky, mean
        zero_float_kernel<<<(BB * FF + 255) / 256, 256, 0, stream>>>(gmean, BB * FF);
        spmm_mean_kernel<<<BN, 256, 0, stream>>>(H, csr, offs, cnt, dinv, bg[br], gmean);

        // g = leaky(gmean @ Wf + bf) -> xc[:, br*128 : br*128+128]
        fc_kernel<<<(BB * DD + 255) / 256, 256, 0, stream>>>(
            gmean, Wf[br], bf_[br], xc, BB, FF, DD, 256, br * DD, 1);
    }

    // head
    fc_kernel<<<(BB * 256 + 255) / 256, 256, 0, stream>>>(xc, W1, b1, t1, BB, 256, 256, 256, 0, 1);
    fc_kernel<<<(BB * 64 + 255) / 256, 256, 0, stream>>>(t1, W2, b2, t2, BB, 256, 64, 64, 0, 1);
    head_out_kernel<<<1, 32, 0, stream>>>(t2, Wo, bo, out);
}
